// DecoderLSTM_81836306858398
// MI455X (gfx1250) — matrix operand order repaired and checked by reading, not compiled, not run on hardware
//
#include <hip/hip_runtime.h>

typedef __bf16 bf16_t;
typedef __attribute__((ext_vector_type(16))) __bf16 v16bf;
typedef __attribute__((ext_vector_type(8)))  __bf16 v8bf;
typedef __attribute__((ext_vector_type(8)))  float  v8f;

#define BATCH  8192
#define HDIM   64
#define TILE_M 16

__device__ __forceinline__ float fast_tanh(float x) {
#if __has_builtin(__builtin_amdgcn_tanhf)
    return __builtin_amdgcn_tanhf(x);     // gfx1250 V_TANH_F32
#else
    float e = __expf(2.0f * x);
    return 1.0f - 2.0f / (e + 1.0f);
#endif
}
// sigmoid(x + b) where hb = 0.5*b :  0.5*tanh(0.5x + 0.5b) + 0.5
__device__ __forceinline__ float sig_b(float x, float hb) {
    return fmaf(0.5f, fast_tanh(fmaf(0.5f, x, hb)), 0.5f);
}

union ABPack { v8bf h[2]; v16bf v; };

// B operand (32x16 bf16): lane gives column n; K = k0 + i, i=0..15 (k0 already
// includes the lane-half*16 K offset). Row-major W[n][k], stride HDIM.
__device__ __forceinline__ v16bf load_b_operand(const float* __restrict__ W, int n, int k0) {
    v16bf b;
#pragma unroll
    for (int i = 0; i < 16; ++i) b[i] = (bf16_t)W[n * HDIM + k0 + i];
    return b;
}

// A operand (16x32 bf16) from LDS row-major h[m][k] (stride HDIM):
// lanes 0-15: V0..7 = K kb+0..15 ; lanes 16-31: K kb+16..31.
__device__ __forceinline__ v16bf load_a_operand(const bf16_t* hbuf, int mrow, int kb, int half) {
    const v8bf* p0 = (const v8bf*)(hbuf + mrow * HDIM + kb + half * 16);
    const v8bf* p1 = (const v8bf*)(hbuf + mrow * HDIM + kb + 8 + half * 16);
    ABPack a;
    a.h[0] = *p0;
    a.h[1] = *p1;
    return a.v;
}

__global__ __launch_bounds__(128, 1) void lstm_decoder_kernel(
    const float* __restrict__ x,      // (B,1,3)
    const float* __restrict__ hidden, // (2,B,64)
    const float* __restrict__ cell,   // (2,B,64)
    const float* __restrict__ Wih0,   // (256,3)
    const float* __restrict__ Whh0,   // (256,64)
    const float* __restrict__ bih0,   // (256)
    const float* __restrict__ bhh0,   // (256)
    const float* __restrict__ Wih1,   // (256,64)
    const float* __restrict__ Whh1,   // (256,64)
    const float* __restrict__ bih1,   // (256)
    const float* __restrict__ bhh1,   // (256)
    const float* __restrict__ Wfc,    // (3,64)
    const float* __restrict__ bfc,    // (3)
    const int*   __restrict__ nstep,
    float* __restrict__ out)          // (B,T,3)
{
    __shared__ __align__(16) bf16_t h0buf[TILE_M * HDIM];
    __shared__ __align__(16) bf16_t h1buf[TILE_M * HDIM];
    __shared__ __align__(16) float  inpbuf[4][TILE_M * 4];

    const int tid  = threadIdx.x;
    const int w    = tid >> 5;        // wave id 0..3
    const int lane = tid & 31;
    const int half = lane >> 4;       // lane-half (K / M split)
    const int col  = lane & 15;       // column within 16-wide tile
    const int base = blockIdx.x * TILE_M;
    const int T    = *nstep;

    // ---- stage initial hidden state into LDS as bf16 ----
    for (int i = tid; i < TILE_M * HDIM; i += 128) {
        int m = i >> 6, k = i & 63;
        h0buf[i] = (bf16_t)hidden[(size_t)(base + m) * HDIM + k];
        h1buf[i] = (bf16_t)hidden[(size_t)(BATCH + base + m) * HDIM + k];
    }
    // ---- initial input x0 into this wave's input buffer ----
    if (col < 3) {
#pragma unroll
        for (int r = 0; r < 8; ++r) {
            int m = r + 8 * half;
            inpbuf[w][m * 4 + col] = x[(size_t)(base + m) * 3 + col];
        }
    }

    // ---- per-lane persistent state: this wave owns hidden columns 16w..16w+15 ----
    const int j = 16 * w + col;       // owned hidden column (C/D tile: lane = column)
    float c0s[8], c1s[8];
#pragma unroll
    for (int r = 0; r < 8; ++r) {
        int m = r + 8 * half;         // C/D tile: row = reg + 8*lane-half
        c0s[r] = cell[(size_t)(base + m) * HDIM + j];
        c1s[r] = cell[(size_t)(BATCH + base + m) * HDIM + j];
    }

    // biases: hb = 0.5*b for sigmoid gates (i,f,o), vb full for tanh gate (g)
    float vb0[4], vb1[4], hb0[4], hb1[4], wc[4][3];
#pragma unroll
    for (int g = 0; g < 4; ++g) {
        int n = 64 * g + j;           // gate index (i/f/g/o block g)
        vb0[g] = bih0[n] + bhh0[n];
        vb1[g] = bih1[n] + bhh1[n];
        hb0[g] = 0.5f * vb0[g];
        hb1[g] = 0.5f * vb1[g];
#pragma unroll
        for (int k = 0; k < 3; ++k) wc[g][k] = Wih0[n * 3 + k];
    }

    // ---- weight B-operands resident in VGPRs (24 x v16bf = 192 VGPRs) ----
    v16bf Bh0[4][2], Bi1[4][2], Bh1[4][2];
#pragma unroll
    for (int g = 0; g < 4; ++g) {
        int n = 64 * g + j;
#pragma unroll
        for (int ko = 0; ko < 2; ++ko) {
            int k0 = ko * 32 + half * 16;
            Bh0[g][ko] = load_b_operand(Whh0, n, k0);
            Bi1[g][ko] = load_b_operand(Wih1, n, k0);
            Bh1[g][ko] = load_b_operand(Whh1, n, k0);
        }
    }
    // Wfc as a zero-padded B operand (columns >= 3 are zero)
    v16bf Bfc[2];
#pragma unroll
    for (int ko = 0; ko < 2; ++ko) {
        int k0 = ko * 32 + half * 16;
        v16bf b;
#pragma unroll
        for (int i = 0; i < 16; ++i)
            b[i] = (col < 3) ? (bf16_t)Wfc[col * HDIM + k0 + i] : (bf16_t)0.0f;
        Bfc[ko] = b;
    }
    float vbfc = (col < 3) ? bfc[col] : 0.0f;

    __syncthreads();

    const int mrow = col;             // A-operand: lane = matrix row
    v16bf a0lo = load_a_operand(h0buf, mrow, 0,  half);
    v16bf a0hi = load_a_operand(h0buf, mrow, 32, half);
    v16bf a1lo = load_a_operand(h1buf, mrow, 0,  half);
    v16bf a1hi = load_a_operand(h1buf, mrow, 32, half);

    for (int t = 0; t < T; ++t) {
        // ================= layer 0 =================
        float xr[8][3];
#pragma unroll
        for (int r = 0; r < 8; ++r) {
            int m = r + 8 * half;
#pragma unroll
            for (int k = 0; k < 3; ++k) xr[r][k] = inpbuf[w][m * 4 + k];
        }
        v8f acc[4];
#pragma unroll
        for (int g = 0; g < 4; ++g) {
            v8f c;
#pragma unroll
            for (int r = 0; r < 8; ++r)   // K=3 input projection in f32 (bias folded later)
                c[r] = xr[r][0] * wc[g][0] + xr[r][1] * wc[g][1] + xr[r][2] * wc[g][2];
            c = __builtin_amdgcn_wmma_f32_16x16x32_bf16(false, a0lo, false, Bh0[g][0], (short)0, c, false, false);
            c = __builtin_amdgcn_wmma_f32_16x16x32_bf16(false, a0hi, false, Bh0[g][1], (short)0, c, false, false);
            acc[g] = c;
        }
#pragma unroll
        for (int r = 0; r < 8; ++r) {
            float ig = sig_b(acc[0][r], hb0[0]);
            float fg = sig_b(acc[1][r], hb0[1]);
            float gg = fast_tanh(acc[2][r] + vb0[2]);
            float og = sig_b(acc[3][r], hb0[3]);
            c0s[r] = fmaf(fg, c0s[r], ig * gg);
            float h = og * fast_tanh(c0s[r]);
            int m = r + 8 * half;
            h0buf[m * HDIM + j] = (bf16_t)h;
        }
        __syncthreads();
        a0lo = load_a_operand(h0buf, mrow, 0,  half);
        a0hi = load_a_operand(h0buf, mrow, 32, half);

        // ================= layer 1 =================
#pragma unroll
        for (int g = 0; g < 4; ++g) {
            v8f c = {};                   // inline-0 accumulator; bias folded into activation
            c = __builtin_amdgcn_wmma_f32_16x16x32_bf16(false, a0lo, false, Bi1[g][0], (short)0, c, false, false);
            c = __builtin_amdgcn_wmma_f32_16x16x32_bf16(false, a0hi, false, Bi1[g][1], (short)0, c, false, false);
            c = __builtin_amdgcn_wmma_f32_16x16x32_bf16(false, a1lo, false, Bh1[g][0], (short)0, c, false, false);
            c = __builtin_amdgcn_wmma_f32_16x16x32_bf16(false, a1hi, false, Bh1[g][1], (short)0, c, false, false);
            acc[g] = c;
        }
#pragma unroll
        for (int r = 0; r < 8; ++r) {
            float ig = sig_b(acc[0][r], hb1[0]);
            float fg = sig_b(acc[1][r], hb1[1]);
            float gg = fast_tanh(acc[2][r] + vb1[2]);
            float og = sig_b(acc[3][r], hb1[3]);
            c1s[r] = fmaf(fg, c1s[r], ig * gg);
            float h = og * fast_tanh(c1s[r]);
            int m = r + 8 * half;
            h1buf[m * HDIM + j] = (bf16_t)h;
        }
        __syncthreads();
        a1lo = load_a_operand(h1buf, mrow, 0,  half);
        a1hi = load_a_operand(h1buf, mrow, 32, half);

        // ============ FC output (every wave; wave 0 stores to global) ============
        v8f oc = {};
        oc = __builtin_amdgcn_wmma_f32_16x16x32_bf16(false, a1lo, false, Bfc[0], (short)0, oc, false, false);
        oc = __builtin_amdgcn_wmma_f32_16x16x32_bf16(false, a1hi, false, Bfc[1], (short)0, oc, false, false);

        if (col < 3) {
#pragma unroll
            for (int r = 0; r < 8; ++r) {
                int m = r + 8 * half;
                float v = oc[r] + vbfc;
                inpbuf[w][m * 4 + col] = v;   // fed back next step (same-wave DS order)
                if (w == 0)
                    out[((size_t)(base + m) * T + t) * 3 + col] = v;
            }
        }
    }
}

extern "C" void kernel_launch(void* const* d_in, const int* in_sizes, int n_in,
                              void* d_out, int out_size, void* d_ws, size_t ws_size,
                              hipStream_t stream) {
    (void)in_sizes; (void)n_in; (void)out_size; (void)d_ws; (void)ws_size;
    lstm_decoder_kernel<<<BATCH / TILE_M, 128, 0, stream>>>(
        (const float*)d_in[0],  (const float*)d_in[1],  (const float*)d_in[2],
        (const float*)d_in[3],  (const float*)d_in[4],  (const float*)d_in[5],
        (const float*)d_in[6],  (const float*)d_in[7],  (const float*)d_in[8],
        (const float*)d_in[9],  (const float*)d_in[10], (const float*)d_in[11],
        (const float*)d_in[12], (const int*)d_in[13],   (float*)d_out);
}
